// N3Block_64295660421295
// MI455X (gfx1250) — matrix operand order repaired and checked by reading, not compiled, not run on hardware
//
#include <hip/hip_runtime.h>
#include <hip/hip_bf16.h>
#include <cstdint>
#include <cstddef>

// ---------------------------------------------------------------------------
// Problem constants (from the reference)
// ---------------------------------------------------------------------------
#define BN    2
#define CDAT  8
#define CFEAT 8
#define HH    160
#define WW    160
#define KSEL  7
#define PSZ   10
#define STR   5
#define N1D   31
#define NSITE (N1D*N1D)          // 961
#define SWIN  15
#define NOFF  (SWIN*SWIN)        // 225
#define PVEC  (CDAT*PSZ*PSZ)     // 800
#define NEGI  (-1.0e9f)
#define UCAND 18                 // union window span for a 4x4 query block
#define NCAND (UCAND*UCAND)      // 324 candidates
#define NTILES 21                // ceil(324/16)
#define KCH   25                 // 800/32 k-chunks
#define QPAD  808                // padded query-patch stride in LDS

typedef __attribute__((ext_vector_type(16))) __bf16 v16bf;
typedef __attribute__((ext_vector_type(8)))  float  v8f;

// ---------------------------------------------------------------------------
// Weight repack: OIHW f32 -> WMMA B-fragment order (bf16, per-lane contiguous)
// ---------------------------------------------------------------------------
__global__ void repack_w(const float* __restrict__ w, unsigned short* __restrict__ packed,
                         int Cout, int Cin, int coutTiles)
{
    const int nkc   = (Cin + 31) >> 5;
    const int total = coutTiles * 9 * nkc * 512;
    const int idx   = blockIdx.x * 256 + threadIdx.x;
    if (idx >= total) return;
    const int j     = idx & 15;
    const int lane  = (idx >> 4) & 31;
    int rest        = idx >> 9;
    const int kc    = rest % nkc; rest /= nkc;
    const int rs    = rest % 9;
    const int ntile = rest / 9;
    const int n     = lane & 15;
    const int khalf = lane >> 4;
    const int c     = kc * 32 + (j < 8 ? khalf * 8 + j : 16 + khalf * 8 + (j - 8));
    const int oc    = ntile * 16 + n;
    float v = 0.f;
    if (oc < Cout && c < Cin) v = w[((size_t)oc * Cin + c) * 9 + rs];
    __bf16 bv = (__bf16)v;
    packed[idx] = *(unsigned short*)&bv;
}

// ---------------------------------------------------------------------------
// 3x3 SAME conv as implicit GEMM on WMMA bf16 (f32 accumulate).
// A-staging: branchless (clamped address + select), coalesced along W.
// ---------------------------------------------------------------------------
__global__ __launch_bounds__(256)
void conv3x3_wmma(const float* __restrict__ in, const unsigned short* __restrict__ wp_u,
                  const float* __restrict__ bias, float* __restrict__ out,
                  int Cin, int Cout, int do_relu)
{
    __shared__ float As[8][16][36];
    const __bf16* wp = (const __bf16*)wp_u;

    const int tid   = threadIdx.x;
    const int wave  = tid >> 5;
    const int lane  = tid & 31;
    const int tile  = blockIdx.x * 8 + wave;     // 10 w-tiles * 160 rows
    const int h     = tile / 10;
    const int w0    = (tile % 10) * 16;
    const int ntile = blockIdx.y;
    const int n0    = ntile * 16;
    const int b     = blockIdx.z;
    const int nkc   = (Cin + 31) >> 5;

    const int m     = lane & 15;
    const int khalf = lane >> 4;
    const int kh8   = khalf * 8;
    const int half  = khalf;      // staging: channel parity
    const int ml    = lane & 15;  // staging: column within tile

    v8f acc = {0.f,0.f,0.f,0.f,0.f,0.f,0.f,0.f};

    for (int rs = 0; rs < 9; ++rs) {
        const int  r    = rs / 3, s = rs % 3;
        const int  row  = h + r - 1;                 // wave-uniform
        const bool rowv = (row >= 0) && (row < HH);  // wave-uniform
        const int  col  = w0 + ml + s - 1;
        const int  colc = min(max(col, 0), WW - 1);
        const bool colok = rowv && (col == colc);

        for (int kc = 0; kc < nkc; ++kc) {
            const int crem  = Cin - kc * 32;
            const int itcnt = min(16, (crem + 1) >> 1);      // uniform
            const float* ipb = in + ((size_t)b * Cin + kc * 32) * (HH * WW)
                                  + (size_t)(rowv ? row : 0) * WW;
            // branchless cooperative stage: lanes 0-15 / 16-31 sweep even/odd channels
            for (int it = 0; it < itcnt; ++it) {
                const int kk = it * 2 + half;
                const int cc = min(kk, crem - 1);
                float v = ipb[(size_t)cc * (HH * WW) + colc];  // unconditional load
                v = (colok && kk == cc) ? v : 0.f;             // v_cndmask, no branch
                As[wave][ml][kk] = v;
            }
            for (int it = itcnt; it < 16; ++it) As[wave][ml][it * 2 + half] = 0.f;
            __syncthreads();

            // A fragment (ISA 16-bit A layout: lane = khalf*16 + m)
            const float* ar = &As[wave][m][0];
            v16bf afrag;
            #pragma unroll
            for (int j = 0; j < 8; ++j) afrag[j]     = (__bf16)ar[kh8 + j];
            #pragma unroll
            for (int j = 0; j < 8; ++j) afrag[8 + j] = (__bf16)ar[16 + kh8 + j];

            // B fragment: pre-packed, one aligned 32B load per lane
            const __bf16* bp = wp + ((((size_t)ntile * 9 + rs) * nkc + kc) << 9) + (lane << 4);
            v16bf bfrag = *(const v16bf*)bp;

            acc = __builtin_amdgcn_wmma_f32_16x16x32_bf16(false, afrag, false, bfrag,
                                                          (short)0, acc, false, false);
            __syncthreads();
        }
    }

    const int oc = n0 + (lane & 15);
    if (oc < Cout) {
        const float bi = bias[oc];
        float rr[8];
        #pragma unroll
        for (int v = 0; v < 8; ++v) {
            float x = acc[v] + bi;
            rr[v] = do_relu ? fmaxf(x, 0.f) : x;
        }
        float* op = out + (((size_t)b * Cout + oc) * HH + h) * WW + w0 + khalf * 8;
        *(float4*)(op)     = make_float4(rr[0], rr[1], rr[2], rr[3]);
        *(float4*)(op + 4) = make_float4(rr[4], rr[5], rr[6], rr[7]);
    }
}

// ---------------------------------------------------------------------------
// Per-site patch stats: yn[q] = ||ye[q]||^2,  temp[q] = exp(mean(lt patch))
// ---------------------------------------------------------------------------
__global__ __launch_bounds__(128)
void patch_stats(const float* __restrict__ xe, const float* __restrict__ lt,
                 float* __restrict__ yn, float* __restrict__ temp)
{
    const int q  = blockIdx.x;
    const int b  = q / NSITE;
    const int rr = q % NSITE;
    const int i1 = rr / N1D, i2 = rr % N1D;
    const int t  = threadIdx.x;

    float s = 0.f;
    for (int e = t; e < PVEC; e += 128) {
        const int c = e / 100, pr = (e % 100) / 10, pc = e % 10;
        const float v = xe[(((size_t)b * CDAT + c) * HH + i1 * STR + pr) * WW + i2 * STR + pc];
        s += v * v;
    }
    float sl = 0.f;
    for (int e = t; e < 100; e += 128)
        sl += lt[((size_t)b * HH + i1 * STR + e / 10) * WW + i2 * STR + e % 10];

    __shared__ float red[128], red2[128];
    red[t] = s; red2[t] = sl; __syncthreads();
    for (int o = 64; o > 0; o >>= 1) {
        if (t < o) { red[t] += red[t + o]; red2[t] += red2[t + o]; }
        __syncthreads();
    }
    if (t == 0) { yn[q] = red[0]; temp[q] = __expf(red2[0] * 0.01f); }
}

// ---------------------------------------------------------------------------
// Windowed patch distances on WMMA.
// Query block = 4x4 sites (M=16); union window = 18x18 candidates (21 N-tiles);
// K = 800 patch dims in 25 bf16 chunks. Epilogue converts dots to logits
// (-D/temp, self-masked) and scatters into each query's own 225-offset row.
// ---------------------------------------------------------------------------
__global__ __launch_bounds__(256)
void wmma_dist(const float* __restrict__ xe, const float* __restrict__ yn,
               const float* __restrict__ temp, float* __restrict__ logits)
{
    __shared__ float qpat[16 * QPAD];   // 16 query patch vectors, f32

    const int qi0 = min(4 * (int)blockIdx.x, N1D - 4);   // clamped 4x4 block
    const int qj0 = min(4 * (int)blockIdx.y, N1D - 4);
    const int b   = blockIdx.z;
    const int bq  = b * NSITE;
    const int t   = threadIdx.x;
    const int wave = t >> 5;
    const int lane = t & 31;
    const int nl    = lane & 15;
    const int khalf = lane >> 4;
    const int kh8   = khalf * 8;

    // stage 16 query patch vectors into LDS
    for (int idx = t; idx < 16 * PVEC; idx += 256) {
        const int m = idx / PVEC, e = idx % PVEC;
        const int i1 = qi0 + (m >> 2), i2 = qj0 + (m & 3);
        const int c = e / 100, pr = (e % 100) / 10, pc = e % 10;
        qpat[m * QPAD + e] =
            xe[(((size_t)b * CDAT + c) * HH + i1 * STR + pr) * WW + i2 * STR + pc];
    }
    __syncthreads();

    const int bmin1 = min(max(qi0 - 7, 0), N1D - SWIN);
    const int bmin2 = min(max(qj0 - 7, 0), N1D - SWIN);

    for (int nt = wave; nt < NTILES; nt += 8) {
        // this lane's candidate (same for B column and D column)
        const int ncand = nt * 16 + nl;
        const int wr = ncand / UCAND, wc = ncand % UCAND;
        const int j1 = bmin1 + wr, j2 = bmin2 + wc;
        const bool candok = (ncand < NCAND) && (j1 <= N1D - 1) && (j2 <= N1D - 1);
        const int j1c = min(j1, N1D - 1), j2c = min(j2, N1D - 1);
        const int cb = ((b * CDAT) * HH + j1c * STR) * WW + j2c * STR;

        v8f acc = {0.f,0.f,0.f,0.f,0.f,0.f,0.f,0.f};
        for (int kch = 0; kch < KCH; ++kch) {
            const int kbase = kch * 32;
            // A fragment from LDS
            const float* ar = &qpat[(size_t)(lane & 15) * QPAD + kbase];
            v16bf afrag;
            #pragma unroll
            for (int j = 0; j < 8; ++j) afrag[j]     = (__bf16)ar[kh8 + j];
            #pragma unroll
            for (int j = 0; j < 8; ++j) afrag[8 + j] = (__bf16)ar[16 + kh8 + j];
            // B fragment: patch elems of this lane's candidate in A's k order
            const int ebase = kbase + kh8;
            v16bf bfrag;
            #pragma unroll
            for (int j = 0; j < 16; ++j) {
                const int e = ebase + (j < 8 ? j : 8 + j);
                const int c = e / 100, pr = (e % 100) / 10, pc = e % 10;
                bfrag[j] = (__bf16)xe[cb + c * (HH * WW) + pr * WW + pc];
            }
            acc = __builtin_amdgcn_wmma_f32_16x16x32_bf16(false, afrag, false, bfrag,
                                                          (short)0, acc, false, false);
        }

        // epilogue: dots -> logits, scattered to each query's offset row
        const float ynnb = yn[bq + j1c * N1D + j2c];
        #pragma unroll
        for (int v = 0; v < 8; ++v) {
            const int m  = v + 8 * khalf;
            const int i1 = qi0 + (m >> 2), i2 = qj0 + (m & 3);
            const int q  = bq + i1 * N1D + i2;
            const int t1 = j1 - min(max(i1 - 7, 0), N1D - SWIN);
            const int t2 = j2 - min(max(i2 - 7, 0), N1D - SWIN);
            if (candok && t1 >= 0 && t1 < SWIN && t2 >= 0 && t2 < SWIN) {
                const float d  = yn[q] + ynnb - 2.f * acc[v];
                float lg = -d / temp[q];
                if (j1 == i1 && j2 == i2) lg = NEGI;
                logits[(size_t)q * NOFF + t1 * SWIN + t2] = lg;
            }
        }
    }
}

// ---------------------------------------------------------------------------
// Per-site: 7 iterative-softmax rounds over precomputed logits + aggregation.
// ---------------------------------------------------------------------------
__global__ __launch_bounds__(256)
void softmax_agg(const float* __restrict__ x_data, const float* __restrict__ logits_g,
                 float* __restrict__ out_p)
{
    __shared__ float ypq[PVEC];
    __shared__ float wmat[NOFF * KSEL];
    __shared__ int   nbase[NOFF];
    __shared__ float red[256];

    const int q  = blockIdx.x;
    const int b  = q / NSITE;
    const int rr = q % NSITE;
    const int i1 = rr / N1D, i2 = rr % N1D;
    const int t  = threadIdx.x;

    for (int e = t; e < PVEC; e += 256) {
        const int c = e / 100, pr = (e % 100) / 10, pc = e % 10;
        ypq[e] = x_data[(((size_t)b * CDAT + c) * HH + i1 * STR + pr) * WW + i2 * STR + pc];
    }

    const int base1 = min(max(i1 - 7, 0), N1D - SWIN);
    const int base2 = min(max(i2 - 7, 0), N1D - SWIN);

    float logit = -3.0e38f;
    if (t < NOFF) {
        logit = logits_g[(size_t)q * NOFF + t];
        const int j1 = base1 + t / SWIN;
        const int j2 = base2 + t % SWIN;
        nbase[t] = ((b * CDAT) * HH + j1 * STR) * WW + j2 * STR;
    }
    __syncthreads();

    for (int k = 0; k < KSEL; ++k) {
        red[t] = (t < NOFF) ? logit : -3.0e38f;
        __syncthreads();
        for (int o = 128; o > 0; o >>= 1) { if (t < o) red[t] = fmaxf(red[t], red[t + o]); __syncthreads(); }
        const float mx = red[0];
        __syncthreads();
        const float ev = (t < NOFF) ? __expf(logit - mx) : 0.f;
        red[t] = ev;
        __syncthreads();
        for (int o = 128; o > 0; o >>= 1) { if (t < o) red[t] += red[t + o]; __syncthreads(); }
        const float sum = red[0];
        __syncthreads();
        const float w = ev / sum;
        if (t < NOFF) wmat[t * KSEL + k] = w;
        logit += log1pf(-fminf(w, 1.f - 1e-6f));
    }
    __syncthreads();

    // aggregation: thread owns e = t, t+256, ... (<= 4 elems), 7 accumulators each
    float z[4][KSEL];
    int   eo[4];
    int   ne = 0;
    for (int e = t; e < PVEC; e += 256) {
        eo[ne] = (e / 100) * (HH * WW) + ((e % 100) / 10) * WW + (e % 10);
        #pragma unroll
        for (int k = 0; k < KSEL; ++k) z[ne][k] = 0.f;
        ++ne;
    }
    for (int off = 0; off < NOFF; ++off) {
        const int nb = nbase[off];
        if (off + 1 < NOFF) __builtin_prefetch(x_data + nbase[off + 1], 0, 0);
        float wv[KSEL];
        #pragma unroll
        for (int k = 0; k < KSEL; ++k) wv[k] = wmat[off * KSEL + k];
        for (int ei = 0; ei < ne; ++ei) {
            const float v = x_data[nb + eo[ei]];
            #pragma unroll
            for (int k = 0; k < KSEL; ++k) z[ei][k] += wv[k] * v;
        }
    }

    const size_t qb = (size_t)q * ((KSEL + 1) * PVEC);
    int ei = 0;
    for (int e = t; e < PVEC; e += 256) {
        const float ypv = ypq[e];
        out_p[qb + e] = ypv;
        #pragma unroll
        for (int k = 0; k < KSEL; ++k) out_p[qb + (size_t)(k + 1) * PVEC + e] = z[ei][k] - ypv;
        ++ei;
    }
}

// ---------------------------------------------------------------------------
// Overlap-add fold as a deterministic gather (no atomics).
// ---------------------------------------------------------------------------
__global__ __launch_bounds__(256)
void fold(const float* __restrict__ out_p, float* __restrict__ out)
{
    const int total = BN * (KSEL + 1) * CDAT * HH * WW;
    const int idx   = blockIdx.x * 256 + threadIdx.x;
    if (idx >= total) return;
    const int w  = idx % WW;
    const int h  = (idx / WW) % HH;
    const int ch = (idx / (HH * WW)) % ((KSEL + 1) * CDAT);
    const int b  = idx / ((KSEL + 1) * CDAT * HH * WW);
    const int g  = ch >> 3, c = ch & 7;

    int i1s[2]; int n1 = 0;
    { const int a = h / STR;
      for (int d = 0; d < 2; ++d) { const int i1 = a - d;
          if (i1 >= 0 && i1 <= N1D - 1 && (h - i1 * STR) <= PSZ - 1) i1s[n1++] = i1; } }
    int i2s[2]; int n2 = 0;
    { const int a = w / STR;
      for (int d = 0; d < 2; ++d) { const int i2 = a - d;
          if (i2 >= 0 && i2 <= N1D - 1 && (w - i2 * STR) <= PSZ - 1) i2s[n2++] = i2; } }

    float acc = 0.f;
    for (int u = 0; u < n1; ++u)
        for (int v = 0; v < n2; ++v) {
            const int i1 = i1s[u], i2 = i2s[v];
            const int pr = h - i1 * STR, pc = w - i2 * STR;
            const size_t qb = ((size_t)((b * N1D + i1) * N1D + i2)) * ((KSEL + 1) * PVEC);
            acc += out_p[qb + (size_t)g * PVEC + c * 100 + pr * 10 + pc];
        }
    out[idx] = acc / (float)(n1 * n2);
}

// ---------------------------------------------------------------------------
// Host orchestration
// ---------------------------------------------------------------------------
extern "C" void kernel_launch(void* const* d_in, const int* in_sizes, int n_in,
                              void* d_out, int out_size, void* d_ws, size_t ws_size,
                              hipStream_t stream)
{
    (void)in_sizes; (void)n_in; (void)out_size; (void)ws_size;

    const float* x_data = (const float*)d_in[0];
    const float* x_feat = (const float*)d_in[1];
    const float* ew1 = (const float*)d_in[2];  const float* eb1 = (const float*)d_in[3];
    const float* ew2 = (const float*)d_in[4];  const float* eb2 = (const float*)d_in[5];
    const float* ew3 = (const float*)d_in[6];  const float* eb3 = (const float*)d_in[7];
    const float* tw1 = (const float*)d_in[8];  const float* tb1 = (const float*)d_in[9];
    const float* tw2 = (const float*)d_in[10]; const float* tb2 = (const float*)d_in[11];
    const float* tw3 = (const float*)d_in[12]; const float* tb3 = (const float*)d_in[13];

    uint8_t* ws = (uint8_t*)d_ws;
    const size_t HB = (size_t)BN * 64 * HH * WW * sizeof(float);   // 13,107,200
    float* h1e = (float*)(ws + 0 * HB);
    float* h1t = (float*)(ws + 1 * HB);
    float* h2e = (float*)(ws + 2 * HB);
    float* h2t = (float*)(ws + 3 * HB);
    float* out_p = (float*)(ws + 0);   // 49.2MB, aliases h* (dead by then)

    size_t o = 4 * HB;
    auto alloc = [&](size_t bytes) -> void* {
        void* p = ws + o; o = (o + bytes + 255) & ~(size_t)255; return p;
    };
    float*          xe    = (float*)alloc((size_t)BN * CDAT * HH * WW * sizeof(float));
    float*          lt    = (float*)alloc((size_t)BN * HH * WW * sizeof(float));
    float*          ynb   = (float*)alloc((size_t)BN * NSITE * sizeof(float));
    float*          tmpb  = (float*)alloc((size_t)BN * NSITE * sizeof(float));
    float*          lgts  = (float*)alloc((size_t)BN * NSITE * NOFF * sizeof(float));
    unsigned short* pe1   = (unsigned short*)alloc(4 * 9 * 1 * 512 * 2);
    unsigned short* pe2   = (unsigned short*)alloc(4 * 9 * 2 * 512 * 2);
    unsigned short* pe3   = (unsigned short*)alloc(1 * 9 * 2 * 512 * 2);
    unsigned short* pt1   = (unsigned short*)alloc(4 * 9 * 1 * 512 * 2);
    unsigned short* pt2   = (unsigned short*)alloc(4 * 9 * 2 * 512 * 2);
    unsigned short* pt3   = (unsigned short*)alloc(1 * 9 * 2 * 512 * 2);

    auto g1 = [](int total) { return dim3((unsigned)((total + 255) / 256)); };

    // weight repack (bf16 WMMA B fragments)
    repack_w<<<g1(4*9*1*512), 256, 0, stream>>>(ew1, pe1, 64,  8, 4);
    repack_w<<<g1(4*9*2*512), 256, 0, stream>>>(ew2, pe2, 64, 64, 4);
    repack_w<<<g1(1*9*2*512), 256, 0, stream>>>(ew3, pe3,  8, 64, 1);
    repack_w<<<g1(4*9*1*512), 256, 0, stream>>>(tw1, pt1, 64,  8, 4);
    repack_w<<<g1(4*9*2*512), 256, 0, stream>>>(tw2, pt2, 64, 64, 4);
    repack_w<<<g1(1*9*2*512), 256, 0, stream>>>(tw3, pt3,  1, 64, 1);

    // CNNs (implicit GEMM on WMMA)
    const dim3 cg(200, 4, BN), cg1(200, 1, BN);
    conv3x3_wmma<<<cg,  256, 0, stream>>>(x_feat, pe1, eb1, h1e,  8, 64, 1);
    conv3x3_wmma<<<cg,  256, 0, stream>>>(h1e,    pe2, eb2, h2e, 64, 64, 1);
    conv3x3_wmma<<<cg1, 256, 0, stream>>>(h2e,    pe3, eb3, xe,  64,  8, 0);
    conv3x3_wmma<<<cg,  256, 0, stream>>>(x_feat, pt1, tb1, h1t,  8, 64, 1);
    conv3x3_wmma<<<cg,  256, 0, stream>>>(h1t,    pt2, tb2, h2t, 64, 64, 1);
    conv3x3_wmma<<<cg1, 256, 0, stream>>>(h2t,    pt3, tb3, lt,  64,  1, 0);

    // stats -> WMMA distance logits -> iterative softmax + aggregation -> fold
    patch_stats<<<dim3(BN * NSITE), 128, 0, stream>>>(xe, lt, ynb, tmpb);
    wmma_dist<<<dim3(8, 8, BN), 256, 0, stream>>>(xe, ynb, tmpb, lgts);
    softmax_agg<<<dim3(BN * NSITE), 256, 0, stream>>>(x_data, lgts, out_p);
    fold<<<g1(BN * 64 * HH * WW), 256, 0, stream>>>(out_p, (float*)d_out);
}
